// Head_42537356100109
// MI455X (gfx1250) — matrix operand order repaired and checked by reading, not compiled, not run on hardware
//
#include <hip/hip_runtime.h>
#include <hip/hip_bf16.h>

// Problem constants (reference: B=16, T=2048, C=1024, HS=128)
#define B_   16
#define T_   2048
#define C_   1024
#define HS_  128
#define BT_  (B_ * T_)

#define QT   128   // query rows per block (kernel 2)
#define KT   64    // key rows per tile   (kernel 2)

typedef __attribute__((ext_vector_type(16))) __bf16 v16bf;
typedef __attribute__((ext_vector_type(8)))  float  v8f;
typedef int v4i __attribute__((vector_size(16)));   // matches builtin param type

union FragAB {
    v16bf v;
    uint4 q[2];
};

// ---- fp32 -> bf16 via native cast (backend picks v_cvt for gfx1250) -------
__device__ __forceinline__ unsigned short f2bf(float f) {
    return __builtin_bit_cast(unsigned short, (__bf16)f);
}
__device__ __forceinline__ unsigned int f2bf2(float lo, float hi) {
    return (unsigned int)f2bf(lo) | ((unsigned int)f2bf(hi) << 16);
}

// ---- async global->LDS (ASYNCcnt) with synchronous fallback ---------------
#if defined(__has_builtin)
#  if __has_builtin(__builtin_amdgcn_global_load_async_to_lds_b128)
#    define HAS_ASYNC_LDS 1
#  endif
#endif
#ifndef HAS_ASYNC_LDS
#  define HAS_ASYNC_LDS 0
#endif

#define AS1 __attribute__((address_space(1)))
#define AS3 __attribute__((address_space(3)))

__device__ __forceinline__ void async_copy16(const void* g, void* l) {
#if HAS_ASYNC_LDS
    __builtin_amdgcn_global_load_async_to_lds_b128(
        (AS1 v4i*)g, (AS3 v4i*)l, 0, 0);
#else
    *(uint4*)l = *(const uint4*)g;
#endif
}

__device__ __forceinline__ void wait_async() {
#if HAS_ASYNC_LDS
#  if defined(__has_builtin) && __has_builtin(__builtin_amdgcn_s_wait_asynccnt)
    __builtin_amdgcn_s_wait_asynccnt(0);
#  else
    asm volatile("s_wait_asynccnt 0x0" ::: "memory");
#  endif
#endif
}

// ---------------------------------------------------------------------------
// Kernel 0a: convert x fp32 -> bf16 once (read 128 MB, write 64 MB).
// ---------------------------------------------------------------------------
__global__ __launch_bounds__(256) void convert_x_kernel(
    const float* __restrict__ x, unsigned short* __restrict__ xb)
{
    const size_t base4 = ((size_t)blockIdx.x * 256 + threadIdx.x) * 4;
    #pragma unroll
    for (int i = 0; i < 4; i++) {
        const size_t idx = (base4 + i) * 4;       // element index
        float4 f = *(const float4*)&x[idx];
        uint2 pk;
        pk.x = f2bf2(f.x, f.y);
        pk.y = f2bf2(f.z, f.w);
        *(uint2*)&xb[idx] = pk;
    }
}

// ---------------------------------------------------------------------------
// Kernel 0b: convert Wq/Wk/Wv fp32 -> bf16; fold 1/sqrt(HS) into Wq.
// ---------------------------------------------------------------------------
__global__ __launch_bounds__(256) void convert_w_kernel(
    const float* __restrict__ Wq, const float* __restrict__ Wk,
    const float* __restrict__ Wv,
    unsigned short* __restrict__ Wqb, unsigned short* __restrict__ Wkb,
    unsigned short* __restrict__ Wvb)
{
    const int m = blockIdx.y;
    const float* src = (m == 0) ? Wq : (m == 1) ? Wk : Wv;
    unsigned short* dst = (m == 0) ? Wqb : (m == 1) ? Wkb : Wvb;
    const float sc = (m == 0) ? 0.08838834764831845f : 1.0f;   // 1/sqrt(HS)
    const size_t idx = ((size_t)blockIdx.x * 256 + threadIdx.x) * 4;
    float4 f = *(const float4*)&src[idx];
    uint2 pk;
    pk.x = f2bf2(f.x * sc, f.y * sc);
    pk.y = f2bf2(f.z * sc, f.w * sc);
    *(uint2*)&dst[idx] = pk;
}

// ---------------------------------------------------------------------------
// Kernel 1: projection  q/k/v = x @ W  (bf16 in, bf16 out) with async
// double-buffered LDS staging.  grid.y selects {Q, K, V}.  K is stored
// TRANSPOSED as (B, HS, T); kernel 2 builds WMMA B-fragments of K^T with
// per-element LDS reads (stride KT).  Q scale already folded into Wq.
// ---------------------------------------------------------------------------
__global__ __launch_bounds__(256) void qkv_proj_kernel(
    const unsigned short* __restrict__ xb,     // (BT, C) bf16
    const unsigned short* __restrict__ Wqb,
    const unsigned short* __restrict__ Wkb,
    const unsigned short* __restrict__ Wvb,
    unsigned short* __restrict__ Qb,           // (BT, HS) bf16 (pre-scaled)
    unsigned short* __restrict__ Ktb,          // (B, HS, T) bf16
    unsigned short* __restrict__ Vb)           // (BT, HS) bf16
{
    __shared__ __align__(16) unsigned short xt[2][128 * 32];   // [row][k]
    __shared__ __align__(16) unsigned short wt[2][32 * 128];   // [k][n]

    const int gy = blockIdx.y;                    // 0=Q 1=K 2=V
    const unsigned short* W = (gy == 0) ? Wqb : (gy == 1) ? Wkb : Wvb;
    const int row0 = blockIdx.x * 128;
    const int tid  = threadIdx.x;
    const int lane = tid & 31;
    const int wave = tid >> 5;
    const int s    = wave * 16;

    auto stage = [&](int buf, int kk) {
        #pragma unroll
        for (int i = 0; i < 2; i++) {             // x tile: 512 x 16B
            int slot = tid + i * 256;
            int r = slot >> 2;                    // 0..127
            int c = (slot & 3) * 8;               // 0..24
            async_copy16(&xb[(size_t)(row0 + r) * C_ + kk + c],
                         &xt[buf][r * 32 + c]);
        }
        #pragma unroll
        for (int i = 0; i < 2; i++) {             // W tile: 512 x 16B
            int slot = tid + i * 256;
            int r = slot >> 4;                    // 0..31
            int c = (slot & 15) * 8;              // 0..120
            async_copy16(&W[(size_t)(kk + r) * HS_ + c],
                         &wt[buf][r * 128 + c]);
        }
    };

    v8f acc[8] = {};                              // 16 x 128 f32 per wave
    stage(0, 0);

    for (int kk = 0; kk < C_; kk += 32) {
        const int buf = (kk >> 5) & 1;
        wait_async();
        __syncthreads();
        if (kk + 32 < C_) stage(buf ^ 1, kk + 32);

        FragAB a;                                 // A: 16x32 of x
        const int arow = s + (lane & 15);
        const int ak   = (lane >> 4) * 8;
        a.q[0] = *(const uint4*)&xt[buf][arow * 32 + ak];
        a.q[1] = *(const uint4*)&xt[buf][arow * 32 + ak + 16];

        #pragma unroll
        for (int nt = 0; nt < 8; nt++) {
            FragAB bm;                            // B: lane = N col, 16 K vals (table order)
            #pragma unroll
            for (int e = 0; e < 16; e++)
                bm.v[e] = __builtin_bit_cast(__bf16, wt[buf][((lane >> 4) * 8 + (e & 7) + (e >> 3) * 16) * 128 + nt * 16 + (lane & 15)]);
            acc[nt] = __builtin_amdgcn_wmma_f32_16x16x32_bf16(
                false, a.v, false, bm.v, (short)0, acc[nt], false, false);
        }
    }

    // ---- epilogue ----
    const int b  = row0 / T_;
    const int t0 = row0 % T_;
    if (gy == 1) {
        // transposed store: Kt[(b*HS + h)*T + t]; 8 consecutive t per lane
        #pragma unroll
        for (int nt = 0; nt < 8; nt++) {
            const int h     = nt * 16 + (lane & 15);
            const int tbase = t0 + s + (lane >> 4) * 8;
            uint4 pk;
            pk.x = f2bf2(acc[nt][0], acc[nt][1]);
            pk.y = f2bf2(acc[nt][2], acc[nt][3]);
            pk.z = f2bf2(acc[nt][4], acc[nt][5]);
            pk.w = f2bf2(acc[nt][6], acc[nt][7]);
            *(uint4*)&Ktb[(size_t)(b * HS_ + h) * T_ + tbase] = pk;
        }
    } else {
        unsigned short* outp = (gy == 0) ? Qb : Vb;
        #pragma unroll
        for (int nt = 0; nt < 8; nt++) {
            const int h     = nt * 16 + (lane & 15);
            const int rbase = row0 + s + (lane >> 4) * 8;
            #pragma unroll
            for (int r = 0; r < 8; r++)
                outp[(size_t)(rbase + r) * HS_ + h] = f2bf(acc[nt][r]);
        }
    }
}

// ---------------------------------------------------------------------------
// Kernel 2: causal flash attention, bf16 WMMA + fp32 online softmax, with
// async double-buffered K^T / V staging.  Block = 8 waves, 128 query rows.
// ---------------------------------------------------------------------------
__global__ __launch_bounds__(256) void attn_kernel(
    const unsigned short* __restrict__ Qb,    // (BT, HS) pre-scaled
    const unsigned short* __restrict__ Ktb,   // (B, HS, T)
    const unsigned short* __restrict__ Vb,    // (BT, HS)
    float* __restrict__ out)                  // (BT, HS) fp32
{
    __shared__ __align__(16) unsigned short kt[2][HS_ * KT];   // K^T [h][t]
    __shared__ __align__(16) unsigned short vt[2][KT * HS_];   // V   [t][h]
    __shared__ __align__(16) unsigned short pt[8][16 * KT];    // per-wave P

    const int b    = blockIdx.y;
    const int q0   = blockIdx.x * QT;
    const int tid  = threadIdx.x;
    const int lane = tid & 31;
    const int wave = tid >> 5;
    const int s    = wave * 16;

    auto stage = [&](int buf, int j0) {
        #pragma unroll
        for (int i = 0; i < 4; i++) {             // K^T tile: 1024 x 16B
            int slot = tid + i * 256;
            int h = slot >> 3;                    // 0..127
            int c = (slot & 7) * 8;               // 0..56
            async_copy16(&Ktb[(size_t)(b * HS_ + h) * T_ + j0 + c],
                         &kt[buf][h * KT + c]);
        }
        #pragma unroll
        for (int i = 0; i < 4; i++) {             // V tile: 1024 x 16B
            int slot = tid + i * 256;
            int t = slot >> 4;                    // 0..63
            int c = (slot & 15) * 8;              // 0..120
            async_copy16(&Vb[(size_t)(b * T_ + j0 + t) * HS_ + c],
                         &vt[buf][t * HS_ + c]);
        }
    };

    // ---- Q A-fragments straight from global in fragment order ----
    FragAB qa[4];
    {
        const unsigned short* qrow =
            &Qb[(size_t)(b * T_ + q0 + s + (lane & 15)) * HS_];
        const int ak = (lane >> 4) * 8;
        #pragma unroll
        for (int kf = 0; kf < 4; kf++) {
            qa[kf].q[0] = *(const uint4*)&qrow[kf * 32 + ak];
            qa[kf].q[1] = *(const uint4*)&qrow[kf * 32 + ak + 16];
        }
    }

    v8f o[8] = {};
    float mi[8], li[8];
    #pragma unroll
    for (int r = 0; r < 8; r++) { mi[r] = -3.0e38f; li[r] = 0.0f; }

    const int jend = q0 + QT;                     // causal upper bound
    stage(0, 0);

    for (int j0 = 0; j0 < jend; j0 += KT) {
        const int buf = (j0 / KT) & 1;
        wait_async();
        __syncthreads();
        if (j0 + KT < jend) stage(buf ^ 1, j0 + KT);

        // ---- S = Q K^T : 4 N-tiles x 4 K-fragments ----
        v8f sacc[4] = {};
        #pragma unroll
        for (int nt = 0; nt < 4; nt++) {
            #pragma unroll
            for (int kf = 0; kf < 4; kf++) {
                FragAB bm;
                #pragma unroll
                for (int e = 0; e < 16; e++)
                    bm.v[e] = __builtin_bit_cast(__bf16, kt[buf][(kf * 32 + (lane >> 4) * 8 + (e & 7) + (e >> 3) * 16) * KT + nt * 16 + (lane & 15)]);
                sacc[nt] = __builtin_amdgcn_wmma_f32_16x16x32_bf16(
                    false, qa[kf].v, false, bm.v, (short)0, sacc[nt], false, false);
            }
        }

        // ---- causal mask (tiles crossing the diagonal for this strip) ----
        if (j0 + KT - 1 > q0 + s) {
            const int qrow_base = q0 + s + (lane >> 4) * 8;
            #pragma unroll
            for (int nt = 0; nt < 4; nt++) {
                const int kg = j0 + nt * 16 + (lane & 15);
                #pragma unroll
                for (int r = 0; r < 8; r++)
                    if (kg > qrow_base + r) sacc[nt][r] = -3.0e38f;
            }
        }

        // ---- online softmax ----
        float alpha[8];
        #pragma unroll
        for (int r = 0; r < 8; r++) {
            float mx = fmaxf(fmaxf(sacc[0][r], sacc[1][r]),
                             fmaxf(sacc[2][r], sacc[3][r]));
            #pragma unroll
            for (int d = 1; d < 16; d <<= 1) mx = fmaxf(mx, __shfl_xor(mx, d, 32));
            const float mn = fmaxf(mi[r], mx);
            alpha[r] = __expf(mi[r] - mn);
            mi[r] = mn;
            float rs = 0.0f;
            #pragma unroll
            for (int nt = 0; nt < 4; nt++) {
                float p = __expf(sacc[nt][r] - mn);
                sacc[nt][r] = p;
                rs += p;
            }
            #pragma unroll
            for (int d = 1; d < 16; d <<= 1) rs += __shfl_xor(rs, d, 32);
            li[r] = li[r] * alpha[r] + rs;
            #pragma unroll
            for (int h = 0; h < 8; h++) o[h][r] *= alpha[r];
        }

        // ---- D-layout P -> per-wave LDS -> A-fragment (no barrier needed) ----
        #pragma unroll
        for (int nt = 0; nt < 4; nt++) {
            const int col = nt * 16 + (lane & 15);
            const int rb  = (lane >> 4) * 8;
            #pragma unroll
            for (int r = 0; r < 8; r++)
                pt[wave][(rb + r) * KT + col] = f2bf(sacc[nt][r]);
        }
        FragAB pa[2];
        {
            const int m  = lane & 15;
            const int ak = (lane >> 4) * 8;
            #pragma unroll
            for (int kf = 0; kf < 2; kf++) {
                pa[kf].q[0] = *(const uint4*)&pt[wave][m * KT + kf * 32 + ak];
                pa[kf].q[1] = *(const uint4*)&pt[wave][m * KT + kf * 32 + ak + 16];
            }
        }

        // ---- O += P @ V : 8 N-tiles x 2 K-fragments ----
        #pragma unroll
        for (int h = 0; h < 8; h++) {
            #pragma unroll
            for (int kf = 0; kf < 2; kf++) {
                FragAB bm;
                #pragma unroll
                for (int e = 0; e < 16; e++)
                    bm.v[e] = __builtin_bit_cast(__bf16, vt[buf][(kf * 32 + (lane >> 4) * 8 + (e & 7) + (e >> 3) * 16) * HS_ + h * 16 + (lane & 15)]);
                o[h] = __builtin_amdgcn_wmma_f32_16x16x32_bf16(
                    false, pa[kf].v, false, bm.v, (short)0, o[h], false, false);
            }
        }
    }

    // ---- epilogue: normalize and store fp32 ----
    const int rbase = q0 + s + (lane >> 4) * 8;
    #pragma unroll
    for (int r = 0; r < 8; r++) {
        const float inv = 1.0f / li[r];
        #pragma unroll
        for (int h = 0; h < 8; h++)
            out[(size_t)(b * T_ + rbase + r) * HS_ + h * 16 + (lane & 15)] =
                o[h][r] * inv;
    }
}

// ---------------------------------------------------------------------------
extern "C" void kernel_launch(void* const* d_in, const int* in_sizes, int n_in,
                              void* d_out, int out_size, void* d_ws, size_t ws_size,
                              hipStream_t stream) {
    (void)in_sizes; (void)n_in; (void)out_size; (void)ws_size;
    const float* x  = (const float*)d_in[0];
    const float* Wq = (const float*)d_in[1];
    const float* Wk = (const float*)d_in[2];
    const float* Wv = (const float*)d_in[3];
    float* out = (float*)d_out;

    // workspace layout (bf16): Qb, Ktb, Vb (8 MB each), xb (64 MB), W*b (256 KB each)
    unsigned short* Qb  = (unsigned short*)d_ws;
    unsigned short* Ktb = Qb  + (size_t)BT_ * HS_;
    unsigned short* Vb  = Ktb + (size_t)BT_ * HS_;
    unsigned short* xb  = Vb  + (size_t)BT_ * HS_;
    unsigned short* Wqb = xb  + (size_t)BT_ * C_;
    unsigned short* Wkb = Wqb + (size_t)C_ * HS_;
    unsigned short* Wvb = Wkb + (size_t)C_ * HS_;

    dim3 blk(256);

    // 0a: x fp32 -> bf16  (BT*C / (256*16) blocks)
    convert_x_kernel<<<dim3((unsigned)((size_t)BT_ * C_ / 4096)), blk, 0, stream>>>(x, xb);
    // 0b: W fp32 -> bf16 (scale folded into Wq)
    convert_w_kernel<<<dim3(C_ * HS_ / 1024, 3), blk, 0, stream>>>(
        Wq, Wk, Wv, Wqb, Wkb, Wvb);

    // 1: projections
    qkv_proj_kernel<<<dim3(BT_ / 128, 3), blk, 0, stream>>>(
        xb, Wqb, Wkb, Wvb, Qb, Ktb, Vb);

    // 2: causal flash attention
    attn_kernel<<<dim3(T_ / QT, B_), blk, 0, stream>>>(Qb, Ktb, Vb, out);
}
